// PosAttBlock_46763603919172
// MI455X (gfx1250) — matrix operand order repaired and checked by reading, not compiled, not run on hardware
//
#include <hip/hip_runtime.h>
#include <hip/hip_bf16.h>

// ---------------------------------------------------------------------------
// Problem constants (reference: B=4, C=512, H=W=64 -> N=4096, MID=64)
// ---------------------------------------------------------------------------
#define BB   4
#define CC   512
#define NN   4096
#define MIDD 64

// LDS leading dims (padded against bank conflicts)
#define LDS_S  65   // f32 score rows   : bank = (row + col) % 64, conflict-free
#define LDS_P  72   // bf16 rows, 144B  : keeps 16B alignment, dword stride 36

typedef __bf16 bf16;
typedef __attribute__((ext_vector_type(16))) __bf16 v16bf;
typedef __attribute__((ext_vector_type(8)))  float  v8f;

#define WMMA_BF16(a, b, c) \
  __builtin_amdgcn_wmma_f32_16x16x32_bf16(false, (a), false, (b), (short)0, (c), false, false)

// ---------------------------------------------------------------------------
// Packed B-operand layout:
// A 32(K) x 16(N) bf16 tile is stored as 512 contiguous elements in per-lane
// fragment order: element B[k][n] lives at  lane*16 + (k%16)  where
// lane = (k/16)*16 + n.  A wave loads its whole fragment with
// ((const v16bf*)tile)[lane]  -> two coalesced global_load_b128 per lane.
// For a (Ktot x Ntot) matrix, tile (kt,nt) starts at (kt*(Ntot/16)+nt)*512.
// ---------------------------------------------------------------------------
__device__ __forceinline__ v16bf load_frag_b_packed(const bf16* __restrict__ tile) {
  int lane = threadIdx.x & 31;
  return ((const v16bf*)tile)[lane];
}

// A-matrix 16x32 bf16, row-major source with leading dim ld (contiguous k order used here, matched by packed B):
// lane L: m = L&15, hi = L>>4; per lane the data is one contiguous 16-element
// run at k = hi*16  -> vectorizes to 2 x b128.
__device__ __forceinline__ v16bf load_frag_a(const bf16* __restrict__ src, int ld) {
  int lane = threadIdx.x & 31;
  int m  = lane & 15;
  int hi = lane >> 4;
  const bf16* row = src + (size_t)m * ld;
  v16bf a;
#pragma unroll
  for (int e = 0; e < 8; ++e) {
    int k0 = hi * 16 + 2 * e;
    a[2 * e]     = row[k0];
    a[2 * e + 1] = row[k0 + 1];
  }
  return a;
}

// C/D f32 16x16: lane L: n = L&15, hi = L>>4; VGPR r -> row m = hi*8 + r.
__device__ __forceinline__ void store_frag_f32(float* __restrict__ dst, int ld, v8f c) {
  int lane = threadIdx.x & 31;
  int n  = lane & 15;
  int hi = lane >> 4;
#pragma unroll
  for (int r = 0; r < 8; ++r) dst[(size_t)(hi * 8 + r) * ld + n] = c[r];
}

// ---------------------------------------------------------------------------
// Stage 1a: weight f32 -> bf16 (row-major; weights are A-operands)
// ---------------------------------------------------------------------------
__global__ void cvt_bf16_kernel(const float* __restrict__ src, bf16* __restrict__ dst, int n) {
  int i = blockIdx.x * blockDim.x + threadIdx.x;
  int stride = gridDim.x * blockDim.x;
  for (; i < n; i += stride) dst[i] = (bf16)src[i];
}

// ---------------------------------------------------------------------------
// Stage 1b: pack x (f32 [B][C][N]) into bf16 packed-B tiles (Ktot=C, Ntot=N).
// ---------------------------------------------------------------------------
__global__ __launch_bounds__(256) void pack_x_kernel(const float* __restrict__ x,
                                                     bf16* __restrict__ xpk) {
  __shared__ float tile[32][65];
  int b  = blockIdx.z;
  int n0 = blockIdx.x * 64;
  int k0 = blockIdx.y * 32;
  const float* xb = x + (size_t)b * CC * NN;
  for (int i = threadIdx.x; i < 32 * 64; i += 256) {
    int r = i >> 6, c = i & 63;
    tile[r][c] = xb[(size_t)(k0 + r) * NN + n0 + c];
  }
  __syncthreads();
  bf16* outb = xpk + (size_t)b * CC * NN;
  if (threadIdx.x < 128) {
    int ntl  = threadIdx.x >> 5;        // n-subtile 0..3
    int lane = threadIdx.x & 31;
    int hi = lane >> 4, dn = lane & 15;
    size_t base = ((size_t)(k0 >> 5) * (NN >> 4) + (size_t)((n0 >> 4) + ntl)) * 512;
    v16bf v;
#pragma unroll
    for (int e = 0; e < 16; ++e) v[e] = (bf16)tile[hi * 16 + e][ntl * 16 + dn];
    ((v16bf*)(outb + base))[lane] = v;
  }
}

// ---------------------------------------------------------------------------
// Stage 2: projection GEMMs.  D[m,n] = sum_k W[m,k] * X[b][k,n] + bias[m]
// out_mode: 0 -> packed-B order with (Kdim=m, Ndim=n)   [K projection]
//           1 -> packed-B order with (Kdim=n, Ndim=m)   [V projection -> Vt]
//           2 -> row-major Out[b][n][m]                 [Q projection -> Qt]
// ---------------------------------------------------------------------------
__global__ __launch_bounds__(256) void gemm_wx_kernel(
    const bf16* __restrict__ W, const float* __restrict__ bias,
    const bf16* __restrict__ Xpk, bf16* __restrict__ Out,
    int M, int N, int K, int out_mode) {
  int b = blockIdx.z;
  const bf16* Xb = Xpk + (size_t)b * K * N;
  int n_base = blockIdx.x * 128;
  int m_base = blockIdx.y * 64;
  int wave = threadIdx.x >> 5;
  int lane = threadIdx.x & 31;
  int m0 = m_base + (wave & 3) * 16;
  int n0 = n_base + (wave >> 2) * 64;

  v8f acc[4] = {};

  for (int k0 = 0; k0 < K; k0 += 32) {
    v16bf a = load_frag_a(W + (size_t)m0 * K + k0, K);
    size_t tbase = ((size_t)(k0 >> 5) * (N >> 4) + (size_t)(n0 >> 4)) * 512;
#pragma unroll
    for (int t = 0; t < 4; ++t) {
      v16bf bf = load_frag_b_packed(Xb + tbase + (size_t)t * 512);
      acc[t] = WMMA_BF16(a, bf, acc[t]);
    }
  }

  int nlo = lane & 15, hi = lane >> 4;
#pragma unroll
  for (int t = 0; t < 4; ++t) {
#pragma unroll
    for (int r = 0; r < 8; ++r) {
      int m = m0 + hi * 8 + r;
      int n = n0 + t * 16 + nlo;
      float v = acc[t][r] + bias[m];
      size_t idx;
      if (out_mode == 0) {
        idx = ((size_t)(m >> 5) * (N >> 4) + (size_t)(n >> 4)) * 512 +
              (size_t)(((m >> 4) & 1) * 256 + (n & 15) * 16 + (m & 15));
        Out[(size_t)b * M * N + idx] = (bf16)v;
      } else if (out_mode == 1) {
        idx = ((size_t)(n >> 5) * (M >> 4) + (size_t)(m >> 4)) * 512 +
              (size_t)(((n >> 4) & 1) * 256 + (m & 15) * 16 + (n & 15));
        Out[(size_t)b * M * N + idx] = (bf16)v;
      } else {
        Out[(size_t)b * N * M + (size_t)n * M + m] = (bf16)v;
      }
    }
  }
}

// ---------------------------------------------------------------------------
// Stage 3: fused flash attention.
//   Qt : [B][N][MID] bf16 row-major (A operand)
//   Kpk: [B] packed-B tiles, Ktot=MID,  Ntot=N   (keys)
//   Vpk: [B] packed-B tiles, Ktot=N,    Ntot=C   (values, transposed)
//   Y  : [B][N][C] f32
// Block = 256 threads (8 waves); query tile 64, key tile 64; each wave owns a
// 64q x 64c accumulator slab.  Softmax is parallelized 64 rows x 4 segments
// across all 256 threads with two-level LDS reductions.
// ---------------------------------------------------------------------------
__global__ __launch_bounds__(256) void attn_kernel(
    const bf16* __restrict__ Qt, const bf16* __restrict__ Kpk,
    const bf16* __restrict__ Vpk, float* __restrict__ Y) {
  __shared__ bf16  sQ[64 * LDS_P];    // query tile (ld = 72)
  __shared__ float sS[64 * LDS_S];    // score tile (ld = 65)
  __shared__ bf16  sP[64 * LDS_P];    // probabilities (ld = 72)
  __shared__ float sM[64], sL[64], sScale[64];
  __shared__ float sPmax[4][64], sPsum[4][64];

  int b  = blockIdx.z;
  int q0 = blockIdx.x * 64;
  const bf16* Qb = Qt  + ((size_t)b * NN + q0) * MIDD;   // contiguous 64x64 tile
  const bf16* Kb = Kpk + (size_t)b * MIDD * NN;
  const bf16* Vb = Vpk + (size_t)b * NN * CC;

  int tid  = threadIdx.x;
  int wave = tid >> 5;
  int lane = tid & 31;
  int c_slab = wave * 64;
  int row = tid & 63;       // softmax row
  int seg = tid >> 6;       // softmax 16-element segment

  // Load Q tile into padded LDS rows via 128-bit copies (8 x uint4 per row).
  for (int i = tid; i < 64 * 8; i += 256) {
    int r = i >> 3, c8 = i & 7;
    ((uint4*)(sQ + r * LDS_P))[c8] = ((const uint4*)(Qb + r * MIDD))[c8];
  }
  if (tid < 64) { sM[tid] = -INFINITY; sL[tid] = 0.0f; }
  __syncthreads();

  v8f acc[4][4] = {};

  for (int jt = 0; jt < NN / 64; ++jt) {
    int j0 = jt * 64;

    // Prefetch next key/value packed tiles (global_prefetch_b8).
    if (jt + 1 < NN / 64) {
      __builtin_prefetch(Kb + (size_t)((j0 >> 4) + 4) * 512, 0, 1);
      __builtin_prefetch(Vb + ((size_t)(((j0 + 64) >> 5)) * (CC >> 4) + (c_slab >> 4)) * 512, 0, 1);
    }

    // ---- S = Qt(64xMID) x K(MIDx64): 16 tiles, 2 per wave -----------------
#pragma unroll
    for (int s = 0; s < 2; ++s) {
      int t  = wave * 2 + s;
      int ti = t >> 2;
      int tj = t & 3;
      v8f sacc = {};
#pragma unroll
      for (int kk = 0; kk < 2; ++kk) {
        v16bf a = load_frag_a(sQ + ti * 16 * LDS_P + kk * 32, LDS_P);
        v16bf bf = load_frag_b_packed(
            Kb + ((size_t)kk * (NN >> 4) + (size_t)((j0 >> 4) + tj)) * 512);
        sacc = WMMA_BF16(a, bf, sacc);
      }
      store_frag_f32(sS + ti * 16 * LDS_S + tj * 16, LDS_S, sacc);
    }
    __syncthreads();

    // ---- online softmax, parallel over 64 rows x 4 segments ---------------
    {  // phase A: per-segment max
      const float* rp = sS + row * LDS_S + seg * 16;
      float mx = rp[0];
#pragma unroll
      for (int j = 1; j < 16; ++j) mx = fmaxf(mx, rp[j]);
      sPmax[seg][row] = mx;
    }
    __syncthreads();
    if (tid < 64) {  // phase B: combine maxes, rescale factor
      float mprev = sM[tid];
      float mx = fmaxf(fmaxf(sPmax[0][tid], sPmax[1][tid]),
                       fmaxf(sPmax[2][tid], sPmax[3][tid]));
      mx = fmaxf(mprev, mx);
      sScale[tid] = __expf(mprev - mx);   // 0 on first tile (mprev = -inf)
      sM[tid] = mx;
    }
    __syncthreads();
    {  // phase C: exp, write P (bf16), per-segment sum
      float m = sM[row];
      const float* rp = sS + row * LDS_S + seg * 16;
      bf16* pp = sP + row * LDS_P + seg * 16;
      float sum = 0.0f;
#pragma unroll
      for (int j = 0; j < 16; ++j) {
        float p = __expf(rp[j] - m);
        pp[j] = (bf16)p;
        sum += p;
      }
      sPsum[seg][row] = sum;
    }
    __syncthreads();
    if (tid < 64) {  // phase D: running denominator
      sL[tid] = sL[tid] * sScale[tid] +
                sPsum[0][tid] + sPsum[1][tid] + sPsum[2][tid] + sPsum[3][tid];
    }

    // ---- rescale accumulators, then acc += P x Vt slab --------------------
    int hi = lane >> 4;
#pragma unroll
    for (int mi = 0; mi < 4; ++mi) {
      float rs[8];
#pragma unroll
      for (int r = 0; r < 8; ++r) rs[r] = sScale[mi * 16 + hi * 8 + r];
#pragma unroll
      for (int ni = 0; ni < 4; ++ni)
#pragma unroll
        for (int r = 0; r < 8; ++r) acc[mi][ni][r] *= rs[r];
    }
#pragma unroll
    for (int kk = 0; kk < 2; ++kk) {
#pragma unroll
      for (int mi = 0; mi < 4; ++mi) {
        v16bf a = load_frag_a(sP + mi * 16 * LDS_P + kk * 32, LDS_P);
#pragma unroll
        for (int ni = 0; ni < 4; ++ni) {
          v16bf bf = load_frag_b_packed(
              Vb + ((size_t)((j0 + kk * 32) >> 5) * (CC >> 4) +
                    (size_t)((c_slab >> 4) + ni)) * 512);
          acc[mi][ni] = WMMA_BF16(a, bf, acc[mi][ni]);
        }
      }
    }
    __syncthreads();
  }

  // ---- epilogue: divide by row sums, store Y[q, c] f32 --------------------
  int hi = lane >> 4, nlo = lane & 15;
#pragma unroll
  for (int mi = 0; mi < 4; ++mi) {
    float inv[8];
#pragma unroll
    for (int r = 0; r < 8; ++r) inv[r] = 1.0f / sL[mi * 16 + hi * 8 + r];
#pragma unroll
    for (int ni = 0; ni < 4; ++ni)
#pragma unroll
      for (int r = 0; r < 8; ++r) {
        int m = mi * 16 + hi * 8 + r;
        int n = c_slab + ni * 16 + nlo;
        Y[((size_t)b * NN + q0 + m) * CC + n] = acc[mi][ni][r] * inv[r];
      }
  }
}

// ---------------------------------------------------------------------------
// Stage 4: out[b][c][n] = alpha * Y[b][n][c] + x[b][c][n]  (LDS transpose)
// ---------------------------------------------------------------------------
__global__ void epilogue_kernel(const float* __restrict__ Y,
                                const float* __restrict__ x,
                                const float* __restrict__ alpha,
                                float* __restrict__ out) {
  __shared__ float tile[32][33];
  int b  = blockIdx.z;
  int n0 = blockIdx.x * 32;
  int c0 = blockIdx.y * 32;
  float a = alpha[0];
#pragma unroll
  for (int i = 0; i < 4; ++i) {
    int n = n0 + threadIdx.y + i * 8;
    tile[threadIdx.y + i * 8][threadIdx.x] =
        Y[((size_t)b * NN + n) * CC + c0 + threadIdx.x];
  }
  __syncthreads();
#pragma unroll
  for (int i = 0; i < 4; ++i) {
    int c = c0 + threadIdx.y + i * 8;
    size_t idx = ((size_t)b * CC + c) * NN + n0 + threadIdx.x;
    out[idx] = a * tile[threadIdx.x][threadIdx.y + i * 8] + x[idx];
  }
}

// ---------------------------------------------------------------------------
// Host-side launch
// ---------------------------------------------------------------------------
extern "C" void kernel_launch(void* const* d_in, const int* in_sizes, int n_in,
                              void* d_out, int out_size, void* d_ws, size_t ws_size,
                              hipStream_t stream) {
  (void)in_sizes; (void)n_in; (void)out_size; (void)ws_size;
  const float* x     = (const float*)d_in[0];
  const float* Wq    = (const float*)d_in[1];
  const float* bq    = (const float*)d_in[2];
  const float* Wk    = (const float*)d_in[3];
  const float* bk    = (const float*)d_in[4];
  const float* Wv    = (const float*)d_in[5];
  const float* bv    = (const float*)d_in[6];
  const float* alpha = (const float*)d_in[7];
  float* out = (float*)d_out;

  // Workspace carve-out (256B aligned): ~71 MB total.
  char* ws = (char*)d_ws;
  size_t off = 0;
  auto alloc = [&](size_t bytes) -> void* {
    void* p = ws + off;
    off += (bytes + 255) & ~(size_t)255;
    return p;
  };
  bf16* xpk = (bf16*)alloc((size_t)BB * CC * NN * sizeof(bf16));   // packed B tiles
  bf16* wqb = (bf16*)alloc((size_t)MIDD * CC * sizeof(bf16));
  bf16* wkb = (bf16*)alloc((size_t)MIDD * CC * sizeof(bf16));
  bf16* wvb = (bf16*)alloc((size_t)CC * CC * sizeof(bf16));
  bf16* Qt  = (bf16*)alloc((size_t)BB * NN * MIDD * sizeof(bf16)); // [B][N][MID]
  bf16* Kpk = (bf16*)alloc((size_t)BB * MIDD * NN * sizeof(bf16)); // packed B tiles
  bf16* Vpk = (bf16*)alloc((size_t)BB * NN * CC * sizeof(bf16));   // packed B tiles
  float* Yb = (float*)alloc((size_t)BB * NN * CC * sizeof(float)); // [B][N][C]

  // Stage 1: weight conversion + x packing.
  cvt_bf16_kernel<<<(MIDD * CC + 255) / 256, 256, 0, stream>>>(Wq, wqb, MIDD * CC);
  cvt_bf16_kernel<<<(MIDD * CC + 255) / 256, 256, 0, stream>>>(Wk, wkb, MIDD * CC);
  cvt_bf16_kernel<<<(CC * CC + 255) / 256, 256, 0, stream>>>(Wv, wvb, CC * CC);
  {
    dim3 blk(256);
    dim3 grd(NN / 64, CC / 32, BB);
    pack_x_kernel<<<grd, blk, 0, stream>>>(x, xpk);
  }

  // Stage 2: projections. WG tile = 64(m) x 128(n).
  {
    dim3 blk(256);
    dim3 gq(NN / 128, MIDD / 64, BB);
    gemm_wx_kernel<<<gq, blk, 0, stream>>>(wqb, bq, xpk, Qt,  MIDD, NN, CC, 2); // Q^T rowmajor
    gemm_wx_kernel<<<gq, blk, 0, stream>>>(wkb, bk, xpk, Kpk, MIDD, NN, CC, 0); // K packed
    dim3 gv(NN / 128, CC / 64, BB);
    gemm_wx_kernel<<<gv, blk, 0, stream>>>(wvb, bv, xpk, Vpk, CC, NN, CC, 1);   // V^T packed
  }

  // Stage 3: fused flash attention.
  {
    dim3 blk(256);
    dim3 grd(NN / 64, 1, BB);
    attn_kernel<<<grd, blk, 0, stream>>>(Qt, Kpk, Vpk, Yb);
  }

  // Stage 4: out = alpha * Y^T + x.
  {
    dim3 blk(32, 8);
    dim3 grd(NN / 32, CC / 32, BB);
    epilogue_kernel<<<grd, blk, 0, stream>>>(Yb, x, alpha, out);
  }
}
